// DeepSeekV2MoEGate_56650618635054
// MI455X (gfx1250) — matrix operand order repaired and checked by reading, not compiled, not run on hardware
//
#include <hip/hip_runtime.h>
#include <hip/hip_bf16.h>
#include <math.h>

#define HIDDEN   5120
#define NEXP     160
#define NGROUP   8
#define EPG      20
#define TOPKG    3
#define TOPK     6
#define RSCALE   16.0f

#define KCH      32               // K per WMMA chunk (bf16 16x16x32)
#define NKCH     (HIDDEN / KCH)   // 160 chunks
#define ROWPAD   20               // padded row stride in dwords (80B: 16B-aligned, conflict-free)
#define NTILE    10               // 160 experts / 16
#define ASZ      (128 * ROWPAD)   // A tile dwords
#define BSZ      (NEXP * ROWPAD)  // B tile dwords
#define BQW      (BSZ / 4 / 8)    // 100 b128 transfers per wave

typedef __attribute__((ext_vector_type(16))) __bf16 v16bf;
typedef __attribute__((ext_vector_type(8)))  float  v8f;
typedef __attribute__((ext_vector_type(4)))  int    v4i;

typedef __attribute__((address_space(1))) v4i* gv4p;
typedef __attribute__((address_space(3))) v4i* lv4p;

union FragU {
    unsigned int u[8];
    v16bf        bf;
};

__device__ __forceinline__ unsigned int pack_bf16x2(float a, float b) {
    union { float f; unsigned int u; } ua, ub;
    ua.f = a; ub.f = b;
    unsigned int ra = ua.u + 0x7FFFu + ((ua.u >> 16) & 1u);
    unsigned int rb = ub.u + 0x7FFFu + ((ub.u >> 16) & 1u);
    return (ra >> 16) | (rb & 0xFFFF0000u);
}

#if __has_builtin(__builtin_amdgcn_global_load_async_to_lds_b128)
#define HAVE_ASYNC_LDS 1
#else
#define HAVE_ASYNC_LDS 0
#endif

__device__ __forceinline__ void wait_asynccnt0() {
#if __has_builtin(__builtin_amdgcn_s_wait_asynccnt)
    __builtin_amdgcn_s_wait_asynccnt(0);
#else
    asm volatile("s_wait_asynccnt 0x0" ::: "memory");
#endif
}

// ---------------------------------------------------------------------------
// Kernel 0: one-time weight pack fp32 -> padded bf16-pair layout
// wsB[(c*160 + e)*ROWPAD + p]
// ---------------------------------------------------------------------------
__global__ __launch_bounds__(256)
void moe_gate_prepw(const float* __restrict__ w, unsigned int* __restrict__ wsB) {
    int id = blockIdx.x * 256 + threadIdx.x;
    if (id >= NKCH * NEXP * 16) return;
    int p = id & 15;
    int e = (id >> 4) % NEXP;
    int c = id / (NEXP * 16);
    const float* src = w + (size_t)e * HIDDEN + c * KCH + p * 2;
    wsB[((size_t)c * NEXP + e) * ROWPAD + p] = pack_bf16x2(src[0], src[1]);
}

// ---------------------------------------------------------------------------
// Kernel 1: logits[T,160] = X[T,5120] * W^T via bf16 WMMA.
// Double-buffered LDS (2-D shared arrays -> ds ops, not flat); async B copy +
// A conversion for chunk c+1 issued BEFORE the chunk-c WMMA burst.
// One barrier per K-chunk.
// ---------------------------------------------------------------------------
__global__ __launch_bounds__(256)
void moe_gate_gemm(const float* __restrict__ x,
                   const unsigned int* __restrict__ wsB,
                   float* __restrict__ logits,
                   int T) {
    __shared__ unsigned int ldsA[2][ASZ];   // 2 x 10240 B
    __shared__ unsigned int ldsB[2][BSZ];   // 2 x 12800 B

    const int tid   = threadIdx.x;
    const int wave  = tid >> 5;
    const int lane  = tid & 31;
    const int row16 = lane & 15;
    const int half  = lane >> 4;
    const int blockBase = blockIdx.x * 128;
    const int m0    = blockBase + wave * 16;

    // issue async copy of B chunk c into ldsB[sel] (100 b128s per wave)
    auto issueB = [&](int c, int sel) {
        const unsigned int* bsrc = wsB + (size_t)c * BSZ;
#pragma unroll
        for (int i = 0; i < 4; ++i) {
            int ofs = i * 32 + lane;        // i<3 provably < BQW
            if (ofs < BQW) {
                int q = wave * BQW + ofs;
#if HAVE_ASYNC_LDS
                __builtin_amdgcn_global_load_async_to_lds_b128(
                    (gv4p)(bsrc + q * 4), (lv4p)(&ldsB[sel][q * 4]), 0, 0);
#else
                uint4 v = ((const uint4*)bsrc)[q];
                *(uint4*)(&ldsB[sel][q * 4]) = v;
#endif
            }
        }
    };

    // convert+store A tile for chunk c into ldsA[sel]
    auto writeA = [&](int c, int sel) {
#pragma unroll
        for (int i = 0; i < 8; ++i) {
            int pidx = tid + i * 256;       // 0..2047
            int r    = pidx >> 4;
            int c2   = pidx & 15;
            int row  = blockBase + r;
            row = row < T ? row : T - 1;    // single v_min; cheap
            const float* src = x + (size_t)row * HIDDEN + c * KCH + c2 * 2;
            ldsA[sel][r * ROWPAD + c2] = pack_bf16x2(src[0], src[1]);
        }
    };

    v8f acc[NTILE];
#pragma unroll
    for (int n = 0; n < NTILE; ++n)
        acc[n] = (v8f){0.f, 0.f, 0.f, 0.f, 0.f, 0.f, 0.f, 0.f};

    // prologue: stage chunk 0
    issueB(0, 0);
    writeA(0, 0);
    wait_asynccnt0();
    __syncthreads();

    for (int c = 0; c < NKCH; ++c) {
        const int cur = c & 1;
        const int nxt = cur ^ 1;

        // stage chunk c+1 (overlaps with this chunk's WMMAs)
        if (c + 1 < NKCH) {
            issueB(c + 1, nxt);
            writeA(c + 1, nxt);
        }

        // ---- A fragment (contiguous k = 16*half + s, matches B) ----
        FragU a;
#pragma unroll
        for (int v = 0; v < 8; ++v) {
            int kp = v + 8 * half;
            a.u[v] = ldsA[cur][(wave * 16 + row16) * ROWPAD + kp];
        }
        // ---- ten B fragments + WMMA burst ----
        FragU b[NTILE];
#pragma unroll
        for (int n = 0; n < NTILE; ++n) {
#pragma unroll
            for (int v = 0; v < 8; ++v)
                b[n].u[v] = ldsB[cur][(n * 16 + row16) * ROWPAD + v + 8 * half];
        }
#pragma unroll
        for (int n = 0; n < NTILE; ++n)
            acc[n] = __builtin_amdgcn_wmma_f32_16x16x32_bf16(
                false, a.bf, false, b[n].bf, (short)0, acc[n], false, false);

        wait_asynccnt0();     // chunk c+1's async copy landed
        __syncthreads();      // + A stores visible; all waves synced
    }

    // ---- epilogue (uniform fast path; T=16384 is tile-divisible) ----
    if (blockBase + 128 <= T) {
#pragma unroll
        for (int n = 0; n < NTILE; ++n) {
            float* dst = logits + (size_t)(m0 + 8 * half) * NEXP + n * 16 + row16;
#pragma unroll
            for (int r = 0; r < 8; ++r)
                dst[(size_t)r * NEXP] = acc[n][r];
        }
    } else {
#pragma unroll
        for (int n = 0; n < NTILE; ++n)
#pragma unroll
            for (int r = 0; r < 8; ++r) {
                int grow = m0 + r + 8 * half;
                if (grow < T)
                    logits[(size_t)grow * NEXP + n * 16 + row16] = acc[n][r];
            }
    }
}

// ---------------------------------------------------------------------------
// Kernel 2: per-token softmax + top-3-of-8 groups + top-6 experts.
// ---------------------------------------------------------------------------
__global__ __launch_bounds__(256)
void moe_gate_route(const float* __restrict__ logits,
                    int* __restrict__ oidx,
                    float* __restrict__ oval,
                    int T) {
    int t = blockIdx.x * blockDim.x + threadIdx.x;
    if (t >= T) return;
    const float* row = logits + (size_t)t * NEXP;

    float gmax[NGROUP];
    float mx = -INFINITY;
#pragma unroll
    for (int g = 0; g < NGROUP; ++g) {
        float gm = -INFINITY;
        for (int j = 0; j < EPG; ++j)
            gm = fmaxf(gm, row[g * EPG + j]);
        gmax[g] = gm;
        mx = fmaxf(mx, gm);
    }

    float se = 0.f;
    for (int i = 0; i < NEXP; ++i)
        se += __expf(row[i] - mx);
    float inv = 1.f / se;

    bool sel[NGROUP];
#pragma unroll
    for (int g = 0; g < NGROUP; ++g) sel[g] = false;
    for (int k = 0; k < TOPKG; ++k) {
        int bg = 0; float bv = -INFINITY;
#pragma unroll
        for (int g = 0; g < NGROUP; ++g)
            if (!sel[g] && gmax[g] > bv) { bv = gmax[g]; bg = g; }
        sel[bg] = true;
    }

    float tv[TOPK];
    int   ti[TOPK];
#pragma unroll
    for (int k = 0; k < TOPK; ++k) { tv[k] = -1.f; ti[k] = 0; }

    for (int g = 0; g < NGROUP; ++g) {
        if (!sel[g]) continue;
        for (int j = 0; j < EPG; ++j) {
            int e = g * EPG + j;
            float s = __expf(row[e] - mx) * inv;
            if (s > tv[TOPK - 1]) {
                int k = TOPK - 1;
                while (k > 0 && s > tv[k - 1]) {
                    tv[k] = tv[k - 1]; ti[k] = ti[k - 1]; --k;
                }
                tv[k] = s; ti[k] = e;
            }
        }
    }

#pragma unroll
    for (int k = 0; k < TOPK; ++k) {
        oidx[(size_t)t * TOPK + k] = ti[k];
        oval[(size_t)t * TOPK + k] = tv[k] * RSCALE;
    }
}

extern "C" void kernel_launch(void* const* d_in, const int* in_sizes, int n_in,
                              void* d_out, int out_size, void* d_ws, size_t ws_size,
                              hipStream_t stream) {
    const float* x = (const float*)d_in[0];   // [4,4096,5120] fp32
    const float* w = (const float*)d_in[1];   // [160,5120]    fp32
    const int T = in_sizes[0] / HIDDEN;       // 16384 tokens

    float*        logits = (float*)d_ws;                                // T*160 f32
    unsigned int* wsB    = (unsigned int*)(logits + (size_t)T * NEXP);  // 512000 dwords

    int*   oidx = (int*)d_out;
    float* oval = (float*)d_out + (size_t)T * TOPK;

    {
        int n = NKCH * NEXP * 16;
        moe_gate_prepw<<<dim3((n + 255) / 256), 256, 0, stream>>>(w, wsB);
    }

    moe_gate_gemm<<<dim3((T + 127) / 128), 256, 0, stream>>>(x, wsB, logits, T);

    moe_gate_route<<<dim3((T + 255) / 256), 256, 0, stream>>>(logits, oidx, oval, T);
}
